// AttentionBlock_70007966925205
// MI455X (gfx1250) — matrix operand order repaired and checked by reading, not compiled, not run on hardware
//
#include <hip/hip_runtime.h>

// ---------------------------------------------------------------------------
// AttentionBlock for MI455X (gfx1250, wave32, WMMA).
// B=8, C=128, H=W=64 -> N=4096, A=16.
// Flash-attention streaming, bf16 WMMA with fp32 accumulation.
// Softmax denominator computed BY the tensor core (P x ones), stabilized by a
// one-time first-tile row max (logits here are bounded ~|10| << fp32 exp range).
// Q/K stored pre-padded to K=32 so every fragment load is branch-free and
// contiguous; V channel groups addressed via immediate offsets off one pointer.
// ---------------------------------------------------------------------------

typedef __attribute__((ext_vector_type(16))) __bf16 v16bf;
typedef __attribute__((ext_vector_type(8)))  __bf16 v8bf;
typedef __attribute__((ext_vector_type(4)))  __bf16 v4bf;
typedef __attribute__((ext_vector_type(8)))  float  v8f;
typedef __attribute__((ext_vector_type(4)))  float  v4f;

#define BB 8
#define CC 128
#define NN 4096
#define AA 16
#define QKS 32                                  // padded a-stride for Q/K

union BFU  { unsigned short u; __bf16 b; };
union V16U { v16bf v; v8bf h[2]; };

// ---------------------------------------------------------------------------
// Kernel 1: 1x1-conv projections.  q,k -> bf16 [B,N,32] (a>=16 zeroed);
// v -> bf16 [B,C,N].  Biases folded here (attn rows sum to 1, bv passes
// through exactly).  Register-tiled: 10 rows x 4 pixels per thread.
// ---------------------------------------------------------------------------
__global__ __launch_bounds__(256) void proj_kernel(
    const float* __restrict__ x,
    const float* __restrict__ Wq, const float* __restrict__ bq,
    const float* __restrict__ Wk, const float* __restrict__ bk,
    const float* __restrict__ Wv, const float* __restrict__ bv,
    __bf16* __restrict__ Qp, __bf16* __restrict__ Kp, __bf16* __restrict__ Vbf)
{
    __shared__ float xt[CC * 64];                   // 32 KB x-tile [c][nl]
    const int b   = blockIdx.x >> 6;                // 64 pixel-tiles per batch
    const int n0  = (blockIdx.x & 63) * 64;
    const int tid = threadIdx.x;

    for (int t = tid; t < CC * 64; t += 256) {
        int c = t >> 6, nl = t & 63;
        xt[t] = x[(size_t)(b * CC + c) * NN + n0 + nl];
    }
    __syncthreads();

    const int rg = tid >> 4;                        // 0..15: row group
    const int pg = tid & 15;                        // 0..15: pixels pg*4..pg*4+3

    // 10 rows per thread: k=0..7 -> Wv row rg+16k; k=8 -> Wq row rg; k=9 -> Wk row rg
    const float* wrow[10];
    #pragma unroll
    for (int k = 0; k < 8; ++k) wrow[k] = Wv + (size_t)(rg + 16 * k) * CC;
    wrow[8] = Wq + (size_t)rg * CC;
    wrow[9] = Wk + (size_t)rg * CC;

    float acc[10][4];
    #pragma unroll
    for (int k = 0; k < 10; ++k)
        #pragma unroll
        for (int p = 0; p < 4; ++p) acc[k][p] = 0.0f;

    for (int c = 0; c < CC; ++c) {
        float xv[4];
        #pragma unroll
        for (int p = 0; p < 4; ++p) xv[p] = xt[c * 64 + pg * 4 + p];
        #pragma unroll
        for (int k = 0; k < 10; ++k) {
            float w = wrow[k][c];
            #pragma unroll
            for (int p = 0; p < 4; ++p) acc[k][p] += w * xv[p];
        }
    }

    // V rows: pack 4 consecutive pixels into one 8-byte store.
    #pragma unroll
    for (int k = 0; k < 8; ++k) {
        int row = rg + 16 * k;
        float bias = bv[row];
        v4bf pk;
        #pragma unroll
        for (int p = 0; p < 4; ++p) pk[p] = (__bf16)(acc[k][p] + bias);
        *(v4bf*)(Vbf + (size_t)(b * CC + row) * NN + n0 + pg * 4) = pk;
    }
    // Q/K rows: [B,N,32] padded layout; zero the a+16 lane of the pad.
    {
        BFU zu; zu.u = 0;
        float biq = bq[rg], bik = bk[rg];
        #pragma unroll
        for (int p = 0; p < 4; ++p) {
            size_t n = (size_t)(b * NN + n0 + pg * 4 + p) * QKS;
            Qp[n + rg]      = (__bf16)(acc[8][p] + biq);
            Qp[n + 16 + rg] = zu.b;
            Kp[n + rg]      = (__bf16)(acc[9][p] + bik);
            Kp[n + 16 + rg] = zu.b;
        }
    }
}

// ---------------------------------------------------------------------------
// Kernel 2: flash attention.  One wave per 16-row query tile (4 waves/block,
// 2048 tiles).  Per 32-column step: 2 WMMA (S) + 9 WMMA (O accumulate + row
// sums via ones-matrix).  No in-loop shuffles, rescales, or branches.
// ---------------------------------------------------------------------------
__global__ __launch_bounds__(128) void attn_kernel(
    const float* __restrict__ x,
    const __bf16* __restrict__ Qp, const __bf16* __restrict__ Kp,
    const __bf16* __restrict__ Vbf, float* __restrict__ out)
{
    __shared__ __bf16 pstage[4][16 * 32];           // per-wave P re-layout tile
    const int lane  = threadIdx.x & 31;
    const int wav   = threadIdx.x >> 5;
    const int tile  = blockIdx.x * 4 + wav;
    const int b     = tile >> 8;                    // 256 i-tiles per batch
    const int ibase = (tile & 255) << 4;
    const int half  = lane >> 4;
    const int l16   = lane & 15;
    const float LOG2E = 1.4426950408889634f;

    BFU onu; onu.u = 0x3F80;                        // bf16 1.0

    // Q tile as A-fragment (16x32): row m=l16; elem e -> K=16h+e.
    // Padding a>=16 is zero in memory.
    V16U qu;
    {
        const __bf16* qrow = Qp + (size_t)(b * NN + ibase + l16) * QKS;
        qu.h[0] = *(const v8bf*)(qrow + 16 * half);
        qu.h[1] = *(const v8bf*)(qrow + 16 * half + 8);
    }
    const v16bf qf = qu.v;

    // All-ones B fragment: Lacc column n = sum_j P[i,j] for every n.
    v16bf onesf;
    #pragma unroll
    for (int e = 0; e < 16; ++e) onesf[e] = onu.b;

    const v8f zf = (v8f)0.0f;

    // Running pointers (advance by 32 columns per iteration).
    const __bf16* kp    = Kp  + (size_t)(b * NN + l16) * QKS + 16 * half;
    const __bf16* vbase = Vbf + (size_t)(b * CC + l16) * NN  + 16 * half;

    // ---- Prepass: row max of first 32-column tile as one-time exp stabilizer.
    float mref2[8];                                 // mref * log2(e)
    {
        v16bf kb0 = *(const v16bf*)kp;
        v16bf kb1 = *(const v16bf*)(kp + 16 * QKS);
        v8f S0 = __builtin_amdgcn_wmma_f32_16x16x32_bf16(false, qf, false, kb0, (short)0, zf, false, false);
        v8f S1 = __builtin_amdgcn_wmma_f32_16x16x32_bf16(false, qf, false, kb1, (short)0, zf, false, false);
        #pragma unroll
        for (int r = 0; r < 8; ++r) {
            float mx = fmaxf(S0[r], S1[r]);
            mx = fmaxf(mx, __shfl_xor(mx, 1, 32));
            mx = fmaxf(mx, __shfl_xor(mx, 2, 32));
            mx = fmaxf(mx, __shfl_xor(mx, 4, 32));
            mx = fmaxf(mx, __shfl_xor(mx, 8, 32));
            mref2[r] = mx * LOG2E;
        }
    }

    v8f O[8];
    #pragma unroll
    for (int cg = 0; cg < 8; ++cg) O[cg] = (v8f)0.0f;
    v8f Lacc = (v8f)0.0f;

    __bf16* pst = pstage[wav];

    #pragma unroll 2
    for (int it = 0; it < NN / 32; ++it) {
        // K B-fragments (32x16): lane = n + 16*(k/16), elem e -> k = 16h+e.
        // a>=16 rows are the zero pad (from memory).
        v16bf kb0 = *(const v16bf*)kp;
        v16bf kb1 = *(const v16bf*)(kp + 16 * QKS);
        kp += 32 * QKS;

        v8f S0 = __builtin_amdgcn_wmma_f32_16x16x32_bf16(false, qf, false, kb0, (short)0, zf, false, false);
        v8f S1 = __builtin_amdgcn_wmma_f32_16x16x32_bf16(false, qf, false, kb1, (short)0, zf, false, false);

        // P = exp2(S*log2e - mref2), staged to LDS as bf16 [16 rows][32 cols].
        // D layout: lane holds column n=l16 of rows m=r+8*half.
        #pragma unroll
        for (int r = 0; r < 8; ++r) {
            int m = r + 8 * half;
            pst[m * 32 + l16]      = (__bf16)__builtin_amdgcn_exp2f(fmaf(S0[r], LOG2E, -mref2[r]));
            pst[m * 32 + 16 + l16] = (__bf16)__builtin_amdgcn_exp2f(fmaf(S1[r], LOG2E, -mref2[r]));
        }
        // Wave-private tile + same-wave LDS in-order: scheduling fence only.
        __builtin_amdgcn_sched_barrier(0);

        // Re-read P as A-fragment: lane row m=l16; K runs [16h..16h+15].
        V16U pu;
        pu.h[0] = *(const v8bf*)(pst + l16 * 32 + 16 * half);
        pu.h[1] = *(const v8bf*)(pst + l16 * 32 + 16 * half + 8);
        __builtin_amdgcn_sched_barrier(0);
        const v16bf pf = pu.v;

        // O += P x V (8 channel groups, immediate offsets) ; Lacc += P x ones.
        #pragma unroll
        for (int cg = 0; cg < 8; ++cg) {
            v16bf vf = *(const v16bf*)(vbase + (size_t)cg * 16 * NN);
            O[cg] = __builtin_amdgcn_wmma_f32_16x16x32_bf16(false, pf, false, vf, (short)0, O[cg], false, false);
        }
        Lacc = __builtin_amdgcn_wmma_f32_16x16x32_bf16(false, pf, false, onesf, (short)0, Lacc, false, false);
        vbase += 32;
    }

    // Epilogue: normalize, add residual, vectorized fp32 store.
    float inv[8];
    #pragma unroll
    for (int r = 0; r < 8; ++r) inv[r] = __builtin_amdgcn_rcpf(Lacc[r]);
    #pragma unroll
    for (int cg = 0; cg < 8; ++cg) {
        size_t base = (size_t)(b * CC + cg * 16 + l16) * NN + ibase + 8 * half;
        v4f x0 = *(const v4f*)(x + base);
        v4f x1 = *(const v4f*)(x + base + 4);
        v4f o0, o1;
        #pragma unroll
        for (int r = 0; r < 4; ++r) {
            o0[r] = O[cg][r]     * inv[r]     + x0[r];
            o1[r] = O[cg][4 + r] * inv[4 + r] + x1[r];
        }
        *(v4f*)(out + base)     = o0;
        *(v4f*)(out + base + 4) = o1;
    }
}

// ---------------------------------------------------------------------------
extern "C" void kernel_launch(void* const* d_in, const int* in_sizes, int n_in,
                              void* d_out, int out_size, void* d_ws, size_t ws_size,
                              hipStream_t stream) {
    const float* x  = (const float*)d_in[0];
    const float* Wq = (const float*)d_in[1];
    const float* bq = (const float*)d_in[2];
    const float* Wk = (const float*)d_in[3];
    const float* bk = (const float*)d_in[4];
    const float* Wv = (const float*)d_in[5];
    const float* bv = (const float*)d_in[6];
    float* out = (float*)d_out;

    // Workspace: Qp (2 MB) + Kp (2 MB) + V (8 MB), all bf16.
    __bf16* Qp  = (__bf16*)d_ws;
    __bf16* Kp  = Qp + (size_t)BB * NN * QKS;
    __bf16* Vbf = Kp + (size_t)BB * NN * QKS;

    proj_kernel<<<BB * (NN / 64), 256, 0, stream>>>(x, Wq, bq, Wk, bk, Wv, bv,
                                                    Qp, Kp, Vbf);
    attn_kernel<<<(BB * (NN / 16)) / 4, 128, 0, stream>>>(x, Qp, Kp, Vbf, out);
}
